// GNNModel_22282290332033
// MI455X (gfx1250) — compile-verified, matrix operand order checked by reading, not run on hardware
//
#include <hip/hip_runtime.h>

typedef __attribute__((ext_vector_type(16))) __bf16 v16bf;
typedef __attribute__((ext_vector_type(8)))  __bf16 v8bf;
typedef __attribute__((ext_vector_type(8)))  float  v8f;

union ABfrag { v16bf v; v8bf h[2]; };

// ---------------------------------------------------------------- utilities
__global__ void zero_f32(float* __restrict__ p, long n) {
  long i = (long)blockIdx.x * blockDim.x + threadIdx.x;
  if (i < n) p[i] = 0.0f;
}

__global__ void degree_kernel(const int* __restrict__ src, const int* __restrict__ dst,
                              float* __restrict__ dout, float* __restrict__ din, int E) {
  int e = blockIdx.x * blockDim.x + threadIdx.x;
  if (e >= E) return;
  unsafeAtomicAdd(&dout[src[e]], 1.0f);
  unsafeAtomicAdd(&din[dst[e]], 1.0f);
}

// in-place: deg -> (deg>0 ? rsqrt(deg) : 0)
__global__ void norm_kernel(float* __restrict__ p, long n) {
  long i = (long)blockIdx.x * blockDim.x + threadIdx.x;
  if (i < n) {
    float d = p[i];
    p[i] = (d > 0.0f) ? rsqrtf(d) : 0.0f;
  }
}

// Convert W[Fin=128][Fout] f32 (row-major, K-major rows) into per-lane bf16
// B-fragments. Fragment (ct, kc, lane) = 16 contiguous bf16:
//   lane<16 : B[kc*32 + {0..7,16..23}][ct*16 + lane]
//   lane>=16: B[kc*32 + {8..15,24..31}][ct*16 + lane-16]
__global__ void wfrag_kernel(const float* __restrict__ W, __bf16* __restrict__ frag, int Fout) {
  int tid = blockIdx.x * blockDim.x + threadIdx.x;
  int total = (Fout / 16) * 4 * 32;
  if (tid >= total) return;
  int lane = tid & 31;
  int kc   = (tid >> 5) & 3;
  int ct   = tid >> 7;
  int n    = ct * 16 + (lane & 15);
  int off  = (lane < 16) ? 0 : 8;
  __bf16* o = &frag[(size_t)tid * 16];
#pragma unroll
  for (int j = 0; j < 8; ++j) {
    int k0 = kc * 32 + off + j;
    int k1 = k0 + 16;
    o[j]     = (__bf16)W[k0 * Fout + n];
    o[8 + j] = (__bf16)W[k1 * Fout + n];
  }
}

// ---------------------------------------------------------------- WMMA GEMM
// out[N,Fout] = scale(in[N,128]) @ W.  Block = 16-node row tile, one wave per
// 16-column tile (blockDim.x = 32 * Fout/16).  Fin fixed at 128 (K loop = 4).
// relu_in==0: x = in * s_out[row]
// relu_in==1: x = relu(in * s_in[row]) * s_out[row]   (s_in >= 0)
__global__ void gemm_kernel(const float* __restrict__ in, const __bf16* __restrict__ wfrag,
                            float* __restrict__ out, const float* __restrict__ s_out,
                            const float* __restrict__ s_in, int relu_in, int Fout) {
  __shared__ __align__(32) __bf16 tile[16 * 128];
  const int node0 = blockIdx.x * 16;

  // Load 16x128 input tile -> bf16 LDS (row-major), fused scaling / relu.
  for (int idx = threadIdx.x * 8; idx < 16 * 128; idx += blockDim.x * 8) {
    int row = idx >> 7, col = idx & 127;
    int node = node0 + row;
    float s = s_out[node];
    if (relu_in) s *= s_in[node];
    const float4* ip = (const float4*)(in + (size_t)node * 128 + col);
    float4 x0 = ip[0], x1 = ip[1];
    float x[8] = {x0.x, x0.y, x0.z, x0.w, x1.x, x1.y, x1.z, x1.w};
    __bf16* tp = &tile[idx];
#pragma unroll
    for (int j = 0; j < 8; ++j) {
      float v = x[j];
      if (relu_in) v = (v > 0.0f) ? v : 0.0f;
      tp[j] = (__bf16)(v * s);
    }
  }
  __syncthreads();

  const int wave = threadIdx.x >> 5;
  const int lane = threadIdx.x & 31;
  const int m    = lane & 15;
  const int off  = (lane < 16) ? 0 : 8;

  v8f acc = {};
#pragma unroll
  for (int kc = 0; kc < 4; ++kc) {
    ABfrag a;
    a.h[0] = *(const v8bf*)&tile[m * 128 + kc * 32 + off];
    a.h[1] = *(const v8bf*)&tile[m * 128 + kc * 32 + off + 16];
    const v16bf* pb = (const v16bf*)&wfrag[(size_t)((wave * 4 + kc) * 32 + lane) * 16];
    acc = __builtin_amdgcn_wmma_f32_16x16x32_bf16(
        /*neg_a=*/false, a.v, /*neg_b=*/false, *pb,
        /*c_mod=*/(short)0, acc, /*reuse_a=*/false, /*reuse_b=*/false);
  }

  // D layout: VGPR r -> M=r (lanes 0-15) / M=r+8 (lanes 16-31), N = lane%16.
  const int n     = wave * 16 + m;
  const int mbase = (lane < 16) ? 0 : 8;
#pragma unroll
  for (int r = 0; r < 8; ++r)
    out[(size_t)(node0 + mbase + r) * Fout + n] = acc[r];
}

// ---------------------------------------------------------------- aggregation
// One wave32 per edge; lane owns F/32 contiguous floats: vector gather of
// t[src] row + hardware f32 atomic adds into agg[dst] row.
template <int F>
__global__ void agg_kernel(const float* __restrict__ t, const int* __restrict__ src,
                           const int* __restrict__ dst, float* __restrict__ agg, int E) {
  int gw   = (int)((blockIdx.x * blockDim.x + threadIdx.x) >> 5);
  int lane = threadIdx.x & 31;
  if (gw >= E) return;
  int s = src[gw], d = dst[gw];
  constexpr int FV = F / 32;
  const float* tp = t + (size_t)s * F + lane * FV;
  float* ap = agg + (size_t)d * F + lane * FV;
  if constexpr (FV == 4) {
    float4 v = *(const float4*)tp;
    unsafeAtomicAdd(ap + 0, v.x);
    unsafeAtomicAdd(ap + 1, v.y);
    unsafeAtomicAdd(ap + 2, v.z);
    unsafeAtomicAdd(ap + 3, v.w);
  } else {
    float2 v = *(const float2*)tp;
    unsafeAtomicAdd(ap + 0, v.x);
    unsafeAtomicAdd(ap + 1, v.y);
  }
}

__global__ void final_scale(float* __restrict__ out, const float* __restrict__ s_in,
                            long n, int F) {
  long i = (long)blockIdx.x * blockDim.x + threadIdx.x;
  if (i < n) out[i] *= s_in[i / F];
}

// ---------------------------------------------------------------- launcher
extern "C" void kernel_launch(void* const* d_in, const int* in_sizes, int n_in,
                              void* d_out, int out_size, void* d_ws, size_t ws_size,
                              hipStream_t stream) {
  const float* feat = (const float*)d_in[0];
  const float* W1   = (const float*)d_in[1];
  const float* W2   = (const float*)d_in[2];
  const float* W3   = (const float*)d_in[3];
  const int*   src  = (const int*)d_in[4];
  const int*   dst  = (const int*)d_in[5];

  const int N  = in_sizes[0] / 128;   // 100000
  const int E  = in_sizes[4];         // 1600000
  const int FO = in_sizes[3] / 128;   // 64

  char* ws = (char*)d_ws;
  size_t off = 0;
  auto carve = [&](size_t bytes) -> void* {
    void* p = ws + off;
    off = (off + bytes + 255) & ~(size_t)255;
    return p;
  };
  float*  tbuf   = (float*)carve((size_t)N * 128 * sizeof(float));
  float*  aggbuf = (float*)carve((size_t)N * 128 * sizeof(float));
  float*  isq    = (float*)carve((size_t)2 * N * sizeof(float));
  __bf16* w1f    = (__bf16*)carve((size_t)128 * 128 * sizeof(__bf16));
  __bf16* w2f    = (__bf16*)carve((size_t)128 * 128 * sizeof(__bf16));
  __bf16* w3f    = (__bf16*)carve((size_t)128 * 64 * sizeof(__bf16));
  float* isq_out = isq;
  float* isq_in  = isq + N;

  const int B = 256;

  // Norms
  zero_f32<<<(2L * N + B - 1) / B, B, 0, stream>>>(isq, 2L * N);
  degree_kernel<<<(E + B - 1) / B, B, 0, stream>>>(src, dst, isq_out, isq_in, E);
  norm_kernel<<<(2L * N + B - 1) / B, B, 0, stream>>>(isq, 2L * N);

  // Weight fragments (f32 -> bf16, WMMA-B layout)
  wfrag_kernel<<<(1024 + B - 1) / B, B, 0, stream>>>(W1, w1f, 128);
  wfrag_kernel<<<(1024 + B - 1) / B, B, 0, stream>>>(W2, w2f, 128);
  wfrag_kernel<<<(512 + B - 1) / B, B, 0, stream>>>(W3, w3f, 64);

  const int nblk   = N / 16;           // 6250 row tiles
  const int aggblk = (E + 7) / 8;      // 8 edges (waves) per 256-thread block

  // Layer 1: t = (feat * isq_out) @ W1 ; agg = scatter-add
  gemm_kernel<<<nblk, 256, 0, stream>>>(feat, w1f, tbuf, isq_out, nullptr, 0, 128);
  zero_f32<<<((long)N * 128 + B - 1) / B, B, 0, stream>>>(aggbuf, (long)N * 128);
  agg_kernel<128><<<aggblk, 256, 0, stream>>>(tbuf, src, dst, aggbuf, E);

  // Layer 2: t = (relu(agg * isq_in) * isq_out) @ W2 ; agg = scatter-add
  gemm_kernel<<<nblk, 256, 0, stream>>>(aggbuf, w2f, tbuf, isq_out, isq_in, 1, 128);
  zero_f32<<<((long)N * 128 + B - 1) / B, B, 0, stream>>>(aggbuf, (long)N * 128);
  agg_kernel<128><<<aggblk, 256, 0, stream>>>(tbuf, src, dst, aggbuf, E);

  // Layer 3: t = (relu(agg * isq_in) * isq_out) @ W3 (Fout=64, 4 waves/block);
  // aggregate straight into d_out, then scale by isq_in (no relu).
  float* outp = (float*)d_out;
  gemm_kernel<<<nblk, 32 * (FO / 16), 0, stream>>>(aggbuf, w3f, tbuf, isq_out, isq_in, 1, FO);
  zero_f32<<<((long)N * FO + B - 1) / B, B, 0, stream>>>(outp, (long)N * FO);
  agg_kernel<64><<<aggblk, 256, 0, stream>>>(tbuf, src, dst, outp, E);
  final_scale<<<((long)N * FO + B - 1) / B, B, 0, stream>>>(outp, isq_in, (long)N * FO, FO);
}
